// AttentionBlock_60060822667522
// MI455X (gfx1250) — matrix operand order repaired and checked by reading, not compiled, not run on hardware
//
#include <hip/hip_runtime.h>

#define HIDDEN 1024
#define NHEADS 16
#define HDIM   64
#define BATCH  2
#define SEQ    2048
#define MROWS  (BATCH * SEQ)   // 4096
#define QKVDIM (3 * HIDDEN)    // 3072

typedef __attribute__((ext_vector_type(16))) __bf16 bf16x16;
typedef __attribute__((ext_vector_type(8)))  __bf16 bf16x8;
typedef __attribute__((ext_vector_type(4)))  __bf16 bf16x4;
typedef __attribute__((ext_vector_type(8)))  float  f32x8;
typedef __attribute__((ext_vector_type(4)))  float  f32x4;
typedef __attribute__((ext_vector_type(4)))  unsigned int u32x4;
typedef __attribute__((ext_vector_type(8)))  int    i32x8;
typedef __attribute__((ext_vector_type(4)))  int    i32x4;

#if defined(__has_builtin)
#if __has_builtin(__builtin_amdgcn_tensor_load_to_lds) && \
    __has_builtin(__builtin_amdgcn_s_wait_tensorcnt)
#define HAVE_TDM 1
#endif
#endif
#ifndef HAVE_TDM
#define HAVE_TDM 0
#endif

// ---------------------------------------------------------------------------
// WMMA operand fragment loaders for v_wmma_f32_16x16x32_bf16 (wave32).
// A matrix 16x32 (MxK), row-major source. Lane l holds row M = l&15;
//   VGPR0..7 <- K [ (l>>4)*16, +16 ) contiguous.
// B matrix 32x16 (KxN), column-major source (contiguous K per column).
//   Lane l holds column N = l&15; K = (l>>4)*16 .. +16 contiguous.
// ---------------------------------------------------------------------------
__device__ __forceinline__ bf16x16 afrag_bf16(const __bf16* row, int kbase, int sel) {
  bf16x8 lo = *(const bf16x8*)(row + kbase + sel * 16);
  bf16x8 hi = *(const bf16x8*)(row + kbase + sel * 16 + 8);
  return __builtin_shufflevector(lo, hi, 0,1,2,3,4,5,6,7,8,9,10,11,12,13,14,15);
}
__device__ __forceinline__ bf16x16 bfrag_bf16(const __bf16* col, int kbase, int sel) {
  return *(const bf16x16*)(col + kbase + sel * 16);
}
__device__ __forceinline__ f32x8 wmma_bf16(bf16x16 a, bf16x16 b, f32x8 c) {
  return __builtin_amdgcn_wmma_f32_16x16x32_bf16(false, a, false, b, (short)0, c,
                                                 false, false);
}

// ---------------------------------------------------------------------------
// TDM: issue a 2D tensor_load_to_lds. Descriptor per CDNA5 ISA ch.8:
// group0 = {count=1, lds_addr, global_addr(57b), type=2}
// group1 = {mask=0, data_size=1(2B), tensor_dim0/1, tile_dim0/1, dim0_stride}
// groups 2/3 (+ trailing group) unused for 2D tiles -> zero.
// This toolchain's builtin is the 6-arg clang-23 form:
//   (u32x4 g0, i32x8 g1, i32x4 g2, i32x4 g3, i32x8 g4, i32 cpol)
// All operands are wave-uniform here (SGPR-materializable).
// ---------------------------------------------------------------------------
#if HAVE_TDM
__device__ __forceinline__ void tdm_load_2d(const void* gsrc, unsigned lds_off,
                                            unsigned tensor_d0, unsigned tensor_d1,
                                            unsigned tile_d0, unsigned tile_d1,
                                            unsigned stride0) {
  unsigned long long ga = (unsigned long long)gsrc;
  u32x4 g0;
  g0[0] = 1u;                                            // count=1, user D#
  g0[1] = lds_off;                                       // LDS byte address
  g0[2] = (unsigned)(ga & 0xffffffffu);                  // global_addr[31:0]
  g0[3] = (unsigned)((ga >> 32) & 0x1ffffffu) | (2u << 30);  // [56:32] | type=2
  i32x8 g1;
  g1[0] = (int)(1u << 16);                               // data_size = 2 bytes
  g1[1] = (int)((tensor_d0 & 0xffffu) << 16);            // tensor_dim0[15:0]
  g1[2] = (int)(((tensor_d0 >> 16) & 0xffffu) | ((tensor_d1 & 0xffffu) << 16));
  g1[3] = (int)(((tensor_d1 >> 16) & 0xffffu) | ((tile_d0 & 0xffffu) << 16));
  g1[4] = (int)(tile_d1 & 0xffffu);                      // tile_dim1, tile_dim2=0
  g1[5] = (int)stride0;                                  // dim0_stride[31:0]
  g1[6] = 0;
  g1[7] = 0;
  i32x4 z4 = {0, 0, 0, 0};
  i32x8 z8 = {0, 0, 0, 0, 0, 0, 0, 0};
  __builtin_amdgcn_tensor_load_to_lds(g0, g1, z4, z4, z8, 0);
}
#endif

// ---------------------------------------------------------------------------
// Kernel 0: one-shot fp32 -> bf16 conversion (vectorized x4).
// ---------------------------------------------------------------------------
__global__ void __launch_bounds__(256)
cvt_bf16_kernel(const float* __restrict__ src, __bf16* __restrict__ dst, int n4) {
  int i = blockIdx.x * 256 + threadIdx.x;
  if (i < n4) {
    f32x4 v = ((const f32x4*)src)[i];
    bf16x4 o;
#pragma unroll
    for (int t = 0; t < 4; ++t) o[t] = (__bf16)v[t];
    ((bf16x4*)dst)[i] = o;
  }
}

// ---------------------------------------------------------------------------
// Kernel 1: qkv = x @ W_qkv^T + b_qkv  (M=4096, K=1024, N=3072), all-bf16 ops.
// 8 waves/block: 2 M-subtiles x 4 N-subtiles; wave tile 32x64
// (2 A frags, 4 B frags, 8 WMMAs per K step). Writes Q,K bf16 [b][h][n][d]
// and V transposed bf16 [b][h][d][n].
// ---------------------------------------------------------------------------
__global__ void __launch_bounds__(256)
qkv_gemm_kernel(const __bf16* __restrict__ Xb, const __bf16* __restrict__ Wq,
                const float* __restrict__ bqkv,
                __bf16* __restrict__ Qb, __bf16* __restrict__ Kb,
                __bf16* __restrict__ Vt)
{
  const int lane = threadIdx.x & 31;
  const int wave = threadIdx.x >> 5;
  const int sel  = lane >> 4;
  const int ln   = lane & 15;
  const int mbase = blockIdx.x * 64 + (wave & 1) * 32;
  const int nbase = blockIdx.y * 256 + (wave >> 1) * 64;

  f32x8 acc[2][4] = {};
  const __bf16* arow0 = Xb + (size_t)(mbase + ln) * HIDDEN;
  const __bf16* arow1 = Xb + (size_t)(mbase + 16 + ln) * HIDDEN;
  const __bf16* brow  = Wq + (size_t)(nbase + ln) * HIDDEN;

  for (int k = 0; k < HIDDEN; k += 32) {
    bf16x16 a0 = afrag_bf16(arow0, k, sel);
    bf16x16 a1 = afrag_bf16(arow1, k, sel);
#pragma unroll
    for (int g = 0; g < 4; ++g) {
      bf16x16 b = bfrag_bf16(brow + (size_t)(g * 16) * HIDDEN, k, sel);
      acc[0][g] = wmma_bf16(a0, b, acc[0][g]);
      acc[1][g] = wmma_bf16(a1, b, acc[1][g]);
    }
  }

#pragma unroll
  for (int g = 0; g < 4; ++g) {
    const int col = nbase + g * 16 + ln;          // 0..3071
    const float bias = bqkv[col];
    const int which = col >> 10;                  // 0=Q 1=K 2=V
    const int c = col & (HIDDEN - 1);
    const int h = c >> 6, d = c & 63;
#pragma unroll
    for (int t = 0; t < 2; ++t) {
#pragma unroll
      for (int r = 0; r < 8; ++r) {
        const int row = mbase + t * 16 + r + 8 * sel;   // 0..4095
        const int b = row >> 11, n = row & (SEQ - 1);
        const __bf16 v = (__bf16)(acc[t][g][r] + bias);
        if (which == 0)      Qb[(((size_t)(b * NHEADS + h)) * SEQ + n) * HDIM + d] = v;
        else if (which == 1) Kb[(((size_t)(b * NHEADS + h)) * SEQ + n) * HDIM + d] = v;
        else                 Vt[(((size_t)(b * NHEADS + h)) * HDIM + d) * SEQ + n] = v;
      }
    }
  }
}

// ---------------------------------------------------------------------------
// Kernel 2: flash attention. 4 waves/block share one (b,h); each wave owns a
// 16-query tile. Per 32-key chunk, wave 0 stages the K tile (32x64) and the
// V^T tile (64x32) into LDS with TDM tensor_load_to_lds (s_wait_tensorcnt),
// then all waves run bf16 WMMAs out of LDS. Online softmax in fp32; P goes
// C-layout -> A-layout through a wave-private LDS bounce (s_wait_dscnt).
// ---------------------------------------------------------------------------
__global__ void __launch_bounds__(128)
attn_kernel(const __bf16* __restrict__ Qb, const __bf16* __restrict__ Kb,
            const __bf16* __restrict__ Vt, __bf16* __restrict__ AO)
{
  __shared__ __align__(64) __bf16 kt[32 * HDIM];     // keys x d (row = key)
  __shared__ __align__(64) __bf16 vt2[HDIM * 32];    // d x keys (row = d)
  __shared__ __align__(64) __bf16 pbuf[4][16 * 32];

  const int lane = threadIdx.x & 31;
  const int wave = threadIdx.x >> 5;
  const int sel  = lane >> 4;
  const int ln   = lane & 15;

  const int bh    = blockIdx.x >> 5;                 // b*16+h
  const int qtile = (blockIdx.x & 31) * 64 + wave * 16;
  const int b = bh >> 4, h = bh & 15;

  const __bf16* Qbh = Qb + (size_t)bh * SEQ * HDIM;
  const __bf16* Kbh = Kb + (size_t)bh * SEQ * HDIM;
  const __bf16* Vbh = Vt + (size_t)bh * HDIM * SEQ;

  const __bf16* qrow = Qbh + (size_t)(qtile + ln) * HDIM;
  const bf16x16 aq0 = afrag_bf16(qrow, 0, sel);
  const bf16x16 aq1 = afrag_bf16(qrow, 32, sel);

  f32x8 o[4] = {};
  float mprev[8], lsum[8];
#pragma unroll
  for (int r = 0; r < 8; ++r) { mprev[r] = -1e30f; lsum[r] = 0.0f; }

  __bf16* pw = &pbuf[wave][0];

  for (int j = 0; j < SEQ; j += 32) {
    __syncthreads();  // previous chunk's LDS reads complete before overwrite
#if HAVE_TDM
    if (wave == 0) {
      tdm_load_2d(Kbh + (size_t)j * HDIM, (unsigned)(size_t)&kt[0],
                  HDIM, SEQ, HDIM, 32, HDIM);
      tdm_load_2d(Vbh + j, (unsigned)(size_t)&vt2[0],
                  SEQ, HDIM, 32, HDIM, SEQ);
      __builtin_amdgcn_s_wait_tensorcnt(0);
    }
#else
    {
      const int t = threadIdx.x;  // 0..127: 128 x 32B covers each 4KB tile
      ((bf16x16*)kt)[t]  = *(const bf16x16*)(Kbh + (size_t)(j + (t >> 2)) * HDIM +
                                             (t & 3) * 16);
      ((bf16x16*)vt2)[t] = *(const bf16x16*)(Vbh + (size_t)(t >> 1) * SEQ + j +
                                             (t & 1) * 16);
    }
#endif
    __syncthreads();

    // ---- scores S[16x32] over keys j..j+31 (K(wmma)=64 split in 2) ----
    f32x8 s0 = {}, s1 = {};
    s0 = wmma_bf16(aq0, bfrag_bf16(kt + ln * HDIM, 0, sel), s0);
    s0 = wmma_bf16(aq1, bfrag_bf16(kt + ln * HDIM, 32, sel), s0);
    s1 = wmma_bf16(aq0, bfrag_bf16(kt + (16 + ln) * HDIM, 0, sel), s1);
    s1 = wmma_bf16(aq1, bfrag_bf16(kt + (16 + ln) * HDIM, 32, sel), s1);

    // ---- online softmax (row r of the tile lives in one VGPR/half) ----
#pragma unroll
    for (int r = 0; r < 8; ++r) {
      const float x0 = s0[r] * 0.125f;   // 1/sqrt(64)
      const float x1 = s1[r] * 0.125f;
      float mx = fmaxf(x0, x1);
      mx = fmaxf(mx, __shfl_xor(mx, 1, 32));
      mx = fmaxf(mx, __shfl_xor(mx, 2, 32));
      mx = fmaxf(mx, __shfl_xor(mx, 4, 32));
      mx = fmaxf(mx, __shfl_xor(mx, 8, 32));
      const float mnew  = fmaxf(mprev[r], mx);
      const float alpha = __expf(mprev[r] - mnew);
      const float p0 = __expf(x0 - mnew);
      const float p1 = __expf(x1 - mnew);
      float ps = p0 + p1;
      ps += __shfl_xor(ps, 1, 32);
      ps += __shfl_xor(ps, 2, 32);
      ps += __shfl_xor(ps, 4, 32);
      ps += __shfl_xor(ps, 8, 32);
      lsum[r]  = lsum[r] * alpha + ps;
      mprev[r] = mnew;
#pragma unroll
      for (int g = 0; g < 4; ++g) o[g][r] *= alpha;

      const int row = r + 8 * sel;
      pw[row * 32 + ln]      = (__bf16)p0;
      pw[row * 32 + 16 + ln] = (__bf16)p1;
    }
    asm volatile("s_wait_dscnt 0" ::: "memory");

    const bf16x16 pf = afrag_bf16(pw + ln * 32, 0, sel);

    // ---- O += P @ V : B operand from V^T tile in LDS ----
#pragma unroll
    for (int g = 0; g < 4; ++g) {
      o[g] = wmma_bf16(pf, bfrag_bf16(vt2 + (g * 16 + ln) * 32, 0, sel), o[g]);
    }
  }

  // ---- normalize + store bf16 attention output [b][n][h*64+d] ----
#pragma unroll
  for (int r = 0; r < 8; ++r) {
    const float inv = 1.0f / lsum[r];
    const int nq = qtile + r + 8 * sel;
#pragma unroll
    for (int g = 0; g < 4; ++g) {
      AO[((size_t)(b * SEQ + nq)) * HIDDEN + h * HDIM + g * 16 + ln] =
          (__bf16)(o[g][r] * inv);
    }
  }
}

// ---------------------------------------------------------------------------
// Kernel 3: out = AO @ W_o^T + b_o  (M=4096, K=1024, N=1024), fp32 out.
// Same 32x64 wave tiling, all-bf16 operands.
// ---------------------------------------------------------------------------
__global__ void __launch_bounds__(256)
out_proj_kernel(const __bf16* __restrict__ AO, const __bf16* __restrict__ Wob,
                const float* __restrict__ bo, float* __restrict__ Out)
{
  const int lane = threadIdx.x & 31;
  const int wave = threadIdx.x >> 5;
  const int sel  = lane >> 4;
  const int ln   = lane & 15;
  const int mbase = blockIdx.x * 64 + (wave & 1) * 32;
  const int nbase = blockIdx.y * 256 + (wave >> 1) * 64;

  f32x8 acc[2][4] = {};
  const __bf16* arow0 = AO + (size_t)(mbase + ln) * HIDDEN;
  const __bf16* arow1 = AO + (size_t)(mbase + 16 + ln) * HIDDEN;
  const __bf16* brow  = Wob + (size_t)(nbase + ln) * HIDDEN;

  for (int k = 0; k < HIDDEN; k += 32) {
    bf16x16 a0 = afrag_bf16(arow0, k, sel);
    bf16x16 a1 = afrag_bf16(arow1, k, sel);
#pragma unroll
    for (int g = 0; g < 4; ++g) {
      bf16x16 b = bfrag_bf16(brow + (size_t)(g * 16) * HIDDEN, k, sel);
      acc[0][g] = wmma_bf16(a0, b, acc[0][g]);
      acc[1][g] = wmma_bf16(a1, b, acc[1][g]);
    }
  }

#pragma unroll
  for (int g = 0; g < 4; ++g) {
    const int col = nbase + g * 16 + ln;
    const float bias = bo[col];
#pragma unroll
    for (int t = 0; t < 2; ++t) {
#pragma unroll
      for (int r = 0; r < 8; ++r) {
        const int row = mbase + t * 16 + r + 8 * sel;
        Out[(size_t)row * HIDDEN + col] = acc[t][g][r] + bias;
      }
    }
  }
}

// ---------------------------------------------------------------------------
extern "C" void kernel_launch(void* const* d_in, const int* in_sizes, int n_in,
                              void* d_out, int out_size, void* d_ws, size_t ws_size,
                              hipStream_t stream) {
  (void)in_sizes; (void)n_in; (void)out_size; (void)ws_size;
  const float* x    = (const float*)d_in[0];
  const float* Wqkv = (const float*)d_in[1];
  const float* bqkv = (const float*)d_in[2];
  const float* Wo   = (const float*)d_in[3];
  const float* bo   = (const float*)d_in[4];
  float* out = (float*)d_out;

  const size_t xsz = (size_t)MROWS * HIDDEN;     // 4 Mi
  const size_t wqs = (size_t)QKVDIM * HIDDEN;    // 3 Mi
  const size_t wos = (size_t)HIDDEN * HIDDEN;    // 1 Mi
  const size_t qsz = (size_t)BATCH * NHEADS * SEQ * HDIM;  // 4 Mi

  __bf16* Xb  = (__bf16*)d_ws;
  __bf16* Wqb = Xb + xsz;
  __bf16* Wob = Wqb + wqs;
  __bf16* Qb  = Wob + wos;
  __bf16* Kb  = Qb + qsz;
  __bf16* Vt  = Kb + qsz;
  __bf16* AO  = Vt + qsz;   // total 24 Mi bf16 = 48 MiB

  cvt_bf16_kernel<<<dim3((int)(xsz / 4 / 256)), 256, 0, stream>>>(x, Xb, (int)(xsz / 4));
  cvt_bf16_kernel<<<dim3((int)(wqs / 4 / 256)), 256, 0, stream>>>(Wqkv, Wqb, (int)(wqs / 4));
  cvt_bf16_kernel<<<dim3((int)(wos / 4 / 256)), 256, 0, stream>>>(Wo, Wob, (int)(wos / 4));

  qkv_gemm_kernel<<<dim3(MROWS / 64, QKVDIM / 256), 256, 0, stream>>>(
      Xb, Wqb, bqkv, Qb, Kb, Vt);
  attn_kernel<<<dim3(BATCH * NHEADS * (SEQ / 64)), 128, 0, stream>>>(
      Qb, Kb, Vt, AO);
  out_proj_kernel<<<dim3(MROWS / 64, HIDDEN / 256), 256, 0, stream>>>(
      AO, Wob, bo, out);
}
